// CoreAttention_54571854463073
// MI455X (gfx1250) — matrix operand order repaired and checked by reading, not compiled, not run on hardware
//
#include <hip/hip_runtime.h>

typedef __attribute__((ext_vector_type(16))) _Float16 v16h;
typedef __attribute__((ext_vector_type(8)))  _Float16 v8h;
typedef __attribute__((ext_vector_type(4)))  _Float16 v4h;
typedef __attribute__((ext_vector_type(8)))  float    v8f;
typedef __attribute__((ext_vector_type(4)))  float    v4f;

#define SEQ 2048
#define BATCH 2
#define HEADS 16
#define HD 128

constexpr int BQ = 128;   // q rows per workgroup (8 waves x 16)
constexpr int TK = 32;    // kv columns per tile
constexpr int KP = 136;   // K tile row pitch in f16 (128 + 8 pad -> no bank conflicts)
constexpr int VP = 40;    // V^T tile row pitch in f16 (32 + 8 pad)
constexpr int PP = 40;    // P tile row pitch in f16

static __device__ __forceinline__ v8f vzero8() {
  v8f z;
#pragma unroll
  for (int i = 0; i < 8; ++i) z[i] = 0.0f;
  return z;
}

// max without fmaxf's NaN-canonicalize overhead. ONLY used on values that are
// already several VALU ops removed from a WMMA destination write: the compiler
// cannot see through inline asm to insert WMMA->VALU hazard NOPs, so direct
// WMMA results must be consumed by compiler-visible ops first.
static __device__ __forceinline__ float fmax_raw(float a, float b) {
  float d;
  asm("v_max_num_f32_e32 %0, %1, %2" : "=v"(d) : "v"(a), "v"(b));
  return d;
}

// Cross-lane XOR shuffle within each 16-lane half via v_permlane16_b32
// (VALU, keeps the contended DS unit free for WMMA operand loads).
static __device__ __forceinline__ float pl16(float v, int selLo, int selHi) {
  int i = __builtin_bit_cast(int, v);
  int r = __builtin_amdgcn_permlane16(i, i, selLo, selHi, false, false);
  return __builtin_bit_cast(float, r);
}

// max across the 16 lanes of each half (rows live per-half in C layout)
static __device__ __forceinline__ float redmax16(float v) {
  v = fmax_raw(v, pl16(v, 0xFEDCBA98, 0x76543210));  // xor 8
  v = fmax_raw(v, pl16(v, 0x32107654, 0xBA98FEDC));  // xor 4
  v = fmax_raw(v, pl16(v, 0x54761032, 0xDCFE98BA));  // xor 2
  v = fmax_raw(v, pl16(v, 0x67452301, 0xEFCDAB89));  // xor 1
  return v;
}

// One 16(q) x 32(kv) attention tile for one wave. MASKED only for the single
// diagonal tile per wave; all earlier tiles take the compare-free path.
template <bool MASKED>
static __device__ __forceinline__ void attn_tile(
    const v16h (&qa)[4], const v16h onesb,
    v8f (&acc)[8], float (&rmax)[8], float (&rsum)[8],
    const _Float16* Ksh, const _Float16* Vsh, _Float16* pw,
    const int kv0, const int qw, const int nn, const int hi) {
  // ---- BMM1: scores = Q * K^T  (two 16x16 tiles, K-dim 128 in 4 chunks) ----
  v8f c0 = vzero8(), c1 = vzero8();
#pragma unroll
  for (int c = 0; c < 4; ++c) {
    v16h kb0 = *(const v16h*)&Ksh[nn * KP + 32 * c + hi * 16];
    c0 = __builtin_amdgcn_wmma_f32_16x16x32_f16(false, qa[c], false, kb0,
                                                (short)0, c0, false, false);
    v16h kb1 = *(const v16h*)&Ksh[(nn + 16) * KP + 32 * c + hi * 16];
    c1 = __builtin_amdgcn_wmma_f32_16x16x32_f16(false, qa[c], false, kb1,
                                                (short)0, c1, false, false);
  }

  // ---- per-row tile max (row m = r + 8*hi, cols striped over 16 lanes).
  // The combine uses fmaxf (compiler-visible) so the WMMA->VALU RAW hazard
  // NOPs are inserted; the butterfly afterwards can use the raw asm max. ----
  float tm[8];
  bool upd = false;
#pragma unroll
  for (int r = 0; r < 8; ++r) {
    float s0 = c0[r], s1 = c1[r];
    if (MASKED) {
      const int qrow = qw + r + hi * 8;
      if (kv0 + nn > qrow)      s0 = -1.0e30f;
      if (kv0 + 16 + nn > qrow) s1 = -1.0e30f;
    }
    tm[r] = redmax16(fmaxf(s0, s1));
    upd   = upd || (tm[r] > rmax[r]);
  }

  if (MASKED || __any(upd)) {  // wave-uniform -> scalar branch, EXEC stays full
    // ---- slow path: max moved; rescale state and accumulators now so no
    // extra per-row state stays live across the WMMA region ----
#pragma unroll
    for (int r = 0; r < 8; ++r) {
      float s0 = c0[r], s1 = c1[r];
      if (MASKED) {
        const int qrow = qw + r + hi * 8;
        if (kv0 + nn > qrow)      s0 = -1.0e30f;
        if (kv0 + 16 + nn > qrow) s1 = -1.0e30f;
      }
      const float nmax  = fmax_raw(rmax[r], tm[r]);
      const float alpha = exp2f(rmax[r] - nmax);
      rmax[r] = nmax;
      rsum[r] *= alpha;
      const float e0 = exp2f(s0 - nmax);
      const float e1 = exp2f(s1 - nmax);
      const int m = r + hi * 8;
      pw[m * PP + nn]      = (_Float16)e0;
      pw[m * PP + 16 + nn] = (_Float16)e1;
#pragma unroll
      for (int dcc = 0; dcc < 8; ++dcc) acc[dcc][r] *= alpha;
    }
  } else {
    // ---- fast path (common): running max unchanged, no rescale needed.
    // s <= tmax <= rmax so exp2(s - rmax) <= 1. ----
#pragma unroll
    for (int r = 0; r < 8; ++r) {
      const float e0 = exp2f(c0[r] - rmax[r]);
      const float e1 = exp2f(c1[r] - rmax[r]);
      const int m = r + hi * 8;
      pw[m * PP + nn]      = (_Float16)e0;
      pw[m * PP + 16 + nn] = (_Float16)e1;
    }
  }

  asm volatile("s_wait_dscnt 0" ::: "memory");  // same-wave cross-lane DS RAW
  v8h plo = *(const v8h*)&pw[nn * PP + hi * 16];       // k = base+0..7
  v8h phi = *(const v8h*)&pw[nn * PP + hi * 16 + 8];   // k = base+8..15
  v16h pa = __builtin_shufflevector(plo, phi,
                                    0, 1, 2, 3, 4, 5, 6, 7,
                                    8, 9, 10, 11, 12, 13, 14, 15);

  // ---- row sums via WMMA against a ones matrix (replaces the DS butterfly);
  // result lands in the same C layout as rsum/rmax. Shares A with PV chain. ----
  v8f csum = __builtin_amdgcn_wmma_f32_16x16x32_f16(false, pa, false, onesb,
                                                    (short)0, vzero8(), true, false);

  // ---- BMM2: out += P * V. All WMMAs share A -> reuse-A hint on all but last ----
#pragma unroll
  for (int dc = 0; dc < 7; ++dc) {
    v16h vb = *(const v16h*)&Vsh[(16 * dc + nn) * VP + hi * 16];
    acc[dc] = __builtin_amdgcn_wmma_f32_16x16x32_f16(false, pa, false, vb,
                                                     (short)0, acc[dc], true, false);
  }
  {
    v16h vb = *(const v16h*)&Vsh[(16 * 7 + nn) * VP + hi * 16];
    acc[7] = __builtin_amdgcn_wmma_f32_16x16x32_f16(false, pa, false, vb,
                                                    (short)0, acc[7], false, false);
  }

  // rsum was pre-scaled in the slow path, so the tail is uniform
#pragma unroll
  for (int r = 0; r < 8; ++r) rsum[r] += csum[r];
}

__global__ __launch_bounds__(256)
void fa_causal_wmma_kernel(const float* __restrict__ Q,
                           const float* __restrict__ K,
                           const float* __restrict__ V,
                           float* __restrict__ O) {
  __shared__ _Float16 Ksh[TK * KP];        // K tile, row-major [kv][d]
  __shared__ _Float16 Vsh[HD * VP];        // V tile, transposed [d][kv]
  __shared__ _Float16 Psh[8 * 16 * PP];    // per-wave P scratch [m][k]

  const int tid  = threadIdx.x;
  // scalarize the wave id so per-wave control flow / addresses stay in SGPRs
  const int wave = __builtin_amdgcn_readfirstlane(tid >> 5);
  const int lane = tid & 31;
  const int nn   = lane & 15;   // n / column index inside 16-lane half
  const int hi   = lane >> 4;   // which 16-lane half

  const int blk = blockIdx.x;
  const int qb  = blk / (BATCH * HEADS);
  const int bh  = blk % (BATCH * HEADS);
  const int b   = bh / HEADS;
  const int h   = bh % HEADS;

  const int q0 = qb * BQ;            // first q row of workgroup
  const int qw = q0 + wave * 16;     // first q row of this wave (scalar)
  const int qi = qw + nn;            // this lane's A-operand row

  const size_t rowStride = (size_t)BATCH * HEADS * HD;  // 4096 floats per seq step
  const size_t headOff   = (size_t)b * HEADS * HD + (size_t)h * HD;

  // fold 1/sqrt(D) and log2(e) into Q so softmax uses exp2
  const float SCL = 0.08838834764831845f * 1.44269504088896340736f;

  // ---- load Q tile (16 x 128) into 4 f16 A-operands, contiguous k = 16*hi + s layout ----
  v16h qa[4];
  {
    const float* qp = Q + (size_t)qi * rowStride + headOff;
#pragma unroll
    for (int c = 0; c < 4; ++c) {
      const float* p = qp + 32 * c + hi * 16;
      v4f f0 = *(const v4f*)(p);        // k = base+0..3
      v4f f1 = *(const v4f*)(p + 4);    // k = base+4..7
      v4f f2 = *(const v4f*)(p + 8);    // k = base+8..11
      v4f f3 = *(const v4f*)(p + 12);   // k = base+12..15
      v16h a;
#pragma unroll
      for (int j = 0; j < 4; ++j) {
        a[j]      = (_Float16)(f0[j] * SCL);
        a[4 + j]  = (_Float16)(f1[j] * SCL);
        a[8 + j]  = (_Float16)(f2[j] * SCL);
        a[12 + j] = (_Float16)(f3[j] * SCL);
      }
      qa[c] = a;
    }
  }

  v16h onesb;
#pragma unroll
  for (int j = 0; j < 16; ++j) onesb[j] = (_Float16)1.0f;

  // ---- flash-attention state ----
  v8f acc[8];
#pragma unroll
  for (int i = 0; i < 8; ++i) acc[i] = vzero8();
  float rmax[8], rsum[8];
#pragma unroll
  for (int r = 0; r < 8; ++r) { rmax[r] = -1.0e30f; rsum[r] = 0.0f; }

  const int nTiles   = (q0 + BQ) / TK;      // causal: tiles up to the WG diagonal
  const int lastTile = (qw + 15) / TK;      // the single tile needing masking (scalar)
  const float* kbase = K + headOff;
  const float* vbase = V + headOff;
  _Float16* pw = &Psh[wave * 16 * PP];

  for (int t = 0; t < nTiles; ++t) {
    const int kv0 = t * TK;
    __syncthreads();  // previous tile fully consumed

    // ---- cooperative stage: K -> Ksh (row-major f16), V -> Vsh (transposed f16) ----
#pragma unroll
    for (int i = 0; i < 4; ++i) {
      const int f   = tid + i * 256;     // float4 slot, 0..1023
      const int row = f >> 5;            // kv row 0..31
      const int c4  = (f & 31) << 2;     // d offset 0..124
      const float* kp = kbase + (size_t)(kv0 + row) * rowStride + c4;
      v4f kf = *(const v4f*)kp;
      v4h khv;
      khv[0] = (_Float16)kf[0]; khv[1] = (_Float16)kf[1];
      khv[2] = (_Float16)kf[2]; khv[3] = (_Float16)kf[3];
      *(v4h*)&Ksh[row * KP + c4] = khv;

      const float* vp = vbase + (size_t)(kv0 + row) * rowStride + c4;
      v4f vf = *(const v4f*)vp;
#pragma unroll
      for (int j = 0; j < 4; ++j)
        Vsh[(c4 + j) * VP + row] = (_Float16)vf[j];
    }
    // prefetch next tile while we compute on this one
    if (t + 1 < nTiles) {
      const size_t nrow = (size_t)(kv0 + TK + lane) * rowStride + (size_t)wave * 16;
      __builtin_prefetch(kbase + nrow, 0, 0);
      __builtin_prefetch(vbase + nrow, 0, 0);
    }
    __syncthreads();  // tile visible to all waves

    if (t > lastTile) continue;  // scalar branch; barriers stay matched

    if (t == lastTile) {
      attn_tile<true >(qa, onesb, acc, rmax, rsum, Ksh, Vsh, pw, kv0, qw, nn, hi);
    } else {
      attn_tile<false>(qa, onesb, acc, rmax, rsum, Ksh, Vsh, pw, kv0, qw, nn, hi);
    }
  }

  // ---- normalize and store: O[s, b, h*D + d] (same flat layout as inputs) ----
  float* ob = O + headOff;
#pragma unroll
  for (int r = 0; r < 8; ++r) {
    const int qrow = qw + r + hi * 8;
    const float inv = 1.0f / rsum[r];
#pragma unroll
    for (int dc = 0; dc < 8; ++dc) {
      ob[(size_t)qrow * rowStride + 16 * dc + nn] = acc[dc][r] * inv;
    }
  }
}

extern "C" void kernel_launch(void* const* d_in, const int* in_sizes, int n_in,
                              void* d_out, int out_size, void* d_ws, size_t ws_size,
                              hipStream_t stream) {
  const float* Q = (const float*)d_in[0];
  const float* K = (const float*)d_in[1];
  const float* V = (const float*)d_in[2];
  // d_in[3] = attention_mask, unused for causal type
  float* O = (float*)d_out;

  dim3 grid((SEQ / BQ) * BATCH * HEADS);  // 512 workgroups
  dim3 block(256);                        // 8 wave32 waves
  fa_causal_wmma_kernel<<<grid, block, 0, stream>>>(Q, K, V, O);
}
